// SimplexAttention_83425444757860
// MI455X (gfx1250) — matrix operand order repaired and checked by reading, not compiled, not run on hardware
//
#include <hip/hip_runtime.h>

// ---------------------------------------------------------------------------
// SimplexAttention for MI455X (gfx1250, wave32, WMMA).
//   Phase 1 (conv_prep): causal conv -> qc,kc as bf16 hi/lo split, fp32 |.|^2,
//                        and V pre-rolled (+1 along seq) and transposed to
//                        [HD,S] bf16 for contiguous WMMA B-operand loads.
//   Phase 2 (attn_fwd):  flash-attention, 16 query rows per wave, 32-col
//                        K tiles; QK^T = 3 bf16 WMMAs (hi*hi+hi*lo+lo*hi),
//                        online softmax with ds_swizzle butterflies,
//                        P relayout via wave-private LDS, PV = 4 bf16 WMMAs.
// ---------------------------------------------------------------------------

typedef __attribute__((ext_vector_type(16))) __bf16       v16bf;
typedef __attribute__((ext_vector_type(8)))  float        v8f;
typedef __attribute__((ext_vector_type(4)))  unsigned int u32x4;
typedef __attribute__((ext_vector_type(8)))  unsigned int u32x8;

static constexpr int B_  = 4;
static constexpr int S_  = 2048;
static constexpr int D_  = 512;
static constexpr int H_  = 8;
static constexpr int HD_ = 64;
static constexpr int C_  = 32;
static constexpr int K_  = 3;
static constexpr int BH_ = B_ * H_;

// float -> bf16 round-to-nearest-even (bit pattern as ushort)
__device__ __forceinline__ unsigned short f2bf(float f) {
  unsigned int u = __float_as_uint(f);
  u += 0x7FFFu + ((u >> 16) & 1u);
  return (unsigned short)(u >> 16);
}

// Build a v16bf WMMA operand from two 16-byte chunks (global or LDS).
__device__ __forceinline__ v16bf ld16bf(const unsigned short* p0,
                                        const unsigned short* p1) {
  u32x4 a = *(const u32x4*)p0;
  u32x4 b = *(const u32x4*)p1;
  u32x8 w = __builtin_shufflevector(a, b, 0, 1, 2, 3, 4, 5, 6, 7);
  return __builtin_bit_cast(v16bf, w);
}

__device__ __forceinline__ v8f ld8f(const float* p) {
  u32x4 a = *(const u32x4*)p;
  u32x4 b = *(const u32x4*)(p + 4);
  return __builtin_bit_cast(v8f, __builtin_shufflevector(a, b, 0, 1, 2, 3, 4, 5, 6, 7));
}

// D = A(16x32 bf16) * B(32x16 bf16) + C(16x16 f32)
#define WMMA_BF16(A, Bm, Cm)                                                   \
  __builtin_amdgcn_wmma_f32_16x16x32_bf16(false, (A), false, (Bm), (short)0,   \
                                          (Cm), false, false)

// ds_swizzle xor-butterfly (group-of-32 mode: imm = (xor<<10)|0x1F)
#define SWZ(v, imm)                                                            \
  __int_as_float(__builtin_amdgcn_ds_swizzle(__float_as_int(v), (imm)))

// ---------------------------------------------------------------------------
// Kernel 1: conv preprocessing + V roll/transpose.
// grid = (B*H, S/64), block = 256
// ---------------------------------------------------------------------------
__global__ __launch_bounds__(256) void conv_prep(
    const float* __restrict__ query, const float* __restrict__ keys,
    const float* __restrict__ vals, const float* __restrict__ conv_w,
    const float* __restrict__ conv_b, unsigned short* __restrict__ qc_hi,
    unsigned short* __restrict__ qc_lo, unsigned short* __restrict__ kc_hi,
    unsigned short* __restrict__ kc_lo, float* __restrict__ q2,
    float* __restrict__ k2, unsigned short* __restrict__ vt) {
  const int bh = blockIdx.x;
  const int b = bh >> 3, h = bh & 7;
  const int s0 = blockIdx.y * 64;
  const int tid = threadIdx.x;

  __shared__ float sW[C_ * HD_ * K_];  // 6144 f32 = 24 KB, [c][d][i]
  __shared__ float sX[2][66 * 67];     // 66 rows (s0-2..s0+63), stride 67 (bank-safe)

  for (int i = tid; i < C_ * HD_ * K_; i += 256)
    sW[i] = conv_w[(size_t)h * C_ * HD_ * K_ + i];

  for (int i = tid; i < 2 * 66 * HD_; i += 256) {
    const int t = i / (66 * HD_);
    const int r = (i / HD_) % 66;
    const int d = i % HD_;
    int ssrc = s0 - 2 + r;             // replicate left pad
    ssrc = ssrc < 0 ? 0 : ssrc;
    const float* src = (t == 0) ? query : keys;
    sX[t][r * 67 + d] = src[((size_t)b * S_ + ssrc) * D_ + h * HD_ + d];
  }
  __syncthreads();

  if (tid < 128) {
    // conv: 64 s-values x 2 tensors, one thread per (tensor, s)
    const int t = tid >> 6;
    const int sl = tid & 63;
    const int sg = s0 + sl;
    unsigned short* chi = (t == 0) ? qc_hi : kc_hi;
    unsigned short* clo = (t == 0) ? qc_lo : kc_lo;
    float* sq = (t == 0) ? q2 : k2;
    const size_t obase = ((size_t)bh * S_ + sg) * C_;
    float ssum = 0.f;
    for (int c = 0; c < C_; ++c) {
      float a = conv_b[h * C_ + c];
      const float* wr = &sW[c * HD_ * K_];
#pragma unroll
      for (int i = 0; i < K_; ++i) {
        const float* xr = &sX[t][(sl + i) * 67];
#pragma unroll
        for (int d = 0; d < HD_; ++d) a = fmaf(xr[d], wr[d * K_ + i], a);
      }
      ssum += a * a;
      const unsigned short hi = f2bf(a);
      const float hif = __uint_as_float(((unsigned int)hi) << 16);
      const unsigned short lo = f2bf(a - hif);
      chi[obase + c] = hi;
      clo[obase + c] = lo;
    }
    sq[(size_t)bh * S_ + sg] = ssum;
  } else {
    // V: roll(+1 over seq) and transpose to [bh][hd][t], bf16
    const int u = tid - 128;
    for (int e = u; e < HD_ * 64; e += 128) {
      const int hd = e & 63;
      const int tl = e >> 6;
      const int tg = s0 + tl;
      const int srcs = (tg + 1) & (S_ - 1);
      const float v = vals[((size_t)b * S_ + srcs) * D_ + h * HD_ + hd];
      vt[((size_t)bh * HD_ + hd) * S_ + tg] = f2bf(v);
    }
  }
}

// ---------------------------------------------------------------------------
// Kernel 2: flash attention. One wave32 per 16 query rows.
// grid = (B*H*S/16)/4, block = 128 (4 waves)
// C/D layout: element g of a lane = row M = g + 8*(lane/16), col N = lane%16.
// A layout: lane holds row (lane%16); 16B chunks at K = 8*half and 16+8*half.
// B layout: lane holds col (lane%16); 16B chunks at K = 8*half and 16+8*half.
// ---------------------------------------------------------------------------
__global__ __launch_bounds__(128) void attn_fwd(
    const unsigned short* __restrict__ qc_hi,
    const unsigned short* __restrict__ qc_lo,
    const unsigned short* __restrict__ kc_hi,
    const unsigned short* __restrict__ kc_lo, const float* __restrict__ q2,
    const float* __restrict__ k2, const unsigned short* __restrict__ vt,
    float* __restrict__ out) {
  const int lane = threadIdx.x & 31;
  const int wv = threadIdx.x >> 5;
  const int hlf = lane >> 4;
  const int ln = lane & 15;
  const int wid = blockIdx.x * 4 + wv;
  const int bh = wid >> 7;           // 128 row-blocks per (b,h)
  const int rb = wid & 127;
  const int b = bh >> 3, h = bh & 7;
  const int row0 = rb << 4;

  // wave-private P tile: 16 rows x 32 cols bf16, row stride 40 ushorts (80B,
  // keeps 16B chunk alignment and staggers LDS banks)
  __shared__ unsigned short pL[4][16 * 40];
  unsigned short* myP = &pL[wv][0];

  const size_t qrow = (size_t)bh * S_ + row0 + ln;
  const unsigned short* qh = qc_hi + qrow * C_;
  const unsigned short* ql = qc_lo + qrow * C_;
  const v16bf aQh = ld16bf(qh + hlf * 8, qh + 16 + hlf * 8);
  const v16bf aQl = ld16bf(ql + hlf * 8, ql + 16 + hlf * 8);
  const v8f myq2 = ld8f(q2 + (size_t)bh * S_ + row0 + hlf * 8);

  v8f acc[4];
  v8f m, l;
#pragma unroll
  for (int g = 0; g < 8; ++g) {
    m[g] = -__builtin_inff();
    l[g] = 0.f;
  }
#pragma unroll
  for (int nt = 0; nt < 4; ++nt)
#pragma unroll
    for (int g = 0; g < 8; ++g) acc[nt][g] = 0.f;

  const int ntiles = (row0 + 16 + 31) >> 5;  // causal: cols 0..row0+15
  for (int ct = 0; ct < ntiles; ++ct) {
    const int t0 = ct << 5;
    if (ct + 1 < ntiles) {  // warm next K tile (global_prefetch_b8)
      const size_t nb = ((size_t)bh * S_ + t0 + 32 + ln) * C_;
      __builtin_prefetch(kc_hi + nb, 0, 1);
      __builtin_prefetch(kc_lo + nb, 0, 1);
    }
    v8f sc0, sc1;
#pragma unroll
    for (int j = 0; j < 2; ++j) {
      const int colr = t0 + j * 16 + ln;
      const size_t kb = ((size_t)bh * S_ + colr) * C_ + hlf * 8;
      const v16bf bKh = ld16bf(kc_hi + kb, kc_hi + kb + 16);
      const v16bf bKl = ld16bf(kc_lo + kb, kc_lo + kb + 16);
      const float k2v = k2[(size_t)bh * S_ + colr];
      v8f cz;
#pragma unroll
      for (int g = 0; g < 8; ++g) cz[g] = 0.f;
      cz = WMMA_BF16(aQh, bKh, cz);  // hi*hi
      cz = WMMA_BF16(aQh, bKl, cz);  // hi*lo
      cz = WMMA_BF16(aQl, bKh, cz);  // lo*hi  (~fp32 qk from bf16 WMMA)
      v8f& sc = j ? sc1 : sc0;
#pragma unroll
      for (int g = 0; g < 8; ++g) {
        float d2 = myq2[g] + k2v - 2.0f * cz[g];
        d2 = fmaxf(d2, 1e-12f);
        const float sv = -__builtin_sqrtf(d2);
        sc[g] = (colr > row0 + g + 8 * hlf) ? -__builtin_inff() : sv;
      }
    }

    // online softmax; butterfly stays inside each 16-lane half (matches M map)
    v8f mn, es;
#pragma unroll
    for (int g = 0; g < 8; ++g) {
      float v = fmaxf(sc0[g], sc1[g]);
      v = fmaxf(v, SWZ(v, 0x041F));
      v = fmaxf(v, SWZ(v, 0x081F));
      v = fmaxf(v, SWZ(v, 0x101F));
      v = fmaxf(v, SWZ(v, 0x201F));
      const float mg = fmaxf(m[g], v);
      mn[g] = mg;
      es[g] = __expf(m[g] - mg);
    }
    v8f p0, p1;
#pragma unroll
    for (int g = 0; g < 8; ++g) {
      p0[g] = __expf(sc0[g] - mn[g]);
      p1[g] = __expf(sc1[g] - mn[g]);
      float ps = p0[g] + p1[g];
      ps += SWZ(ps, 0x041F);
      ps += SWZ(ps, 0x081F);
      ps += SWZ(ps, 0x101F);
      ps += SWZ(ps, 0x201F);
      l[g] = l[g] * es[g] + ps;
      m[g] = mn[g];
    }
#pragma unroll
    for (int nt = 0; nt < 4; ++nt)
#pragma unroll
      for (int g = 0; g < 8; ++g) acc[nt][g] *= es[g];

    // P: D-layout -> LDS row-major -> A-layout (DS ops are in-order per wave)
#pragma unroll
    for (int g = 0; g < 8; ++g) {
      const int M = g + 8 * hlf;
      myP[M * 40 + ln] = f2bf(p0[g]);
      myP[M * 40 + 16 + ln] = f2bf(p1[g]);
    }
    __asm__ volatile("" ::: "memory");
    const unsigned short* pr = myP + ln * 40;
    const v16bf aP = ld16bf(pr + hlf * 8, pr + 16 + hlf * 8);

#pragma unroll
    for (int nt = 0; nt < 4; ++nt) {
      const unsigned short* vp =
          vt + ((size_t)bh * HD_ + nt * 16 + ln) * S_ + t0 + hlf * 8;
      const v16bf bV = ld16bf(vp, vp + 16);
      acc[nt] = WMMA_BF16(aP, bV, acc[nt]);
    }
  }

  // epilogue: normalize and scatter to [B,S,D]
#pragma unroll
  for (int g = 0; g < 8; ++g) {
    const float il = 1.0f / l[g];
    const size_t ob = ((size_t)b * S_ + row0 + g + 8 * hlf) * D_ + h * HD_ + ln;
#pragma unroll
    for (int nt = 0; nt < 4; ++nt) out[ob + nt * 16] = acc[nt][g] * il;
  }
}

// ---------------------------------------------------------------------------
extern "C" void kernel_launch(void* const* d_in, const int* in_sizes, int n_in,
                              void* d_out, int out_size, void* d_ws,
                              size_t ws_size, hipStream_t stream) {
  (void)in_sizes; (void)n_in; (void)out_size; (void)ws_size;
  const float* query = (const float*)d_in[0];
  const float* keys = (const float*)d_in[1];
  const float* vals = (const float*)d_in[2];
  // d_in[3] = mask (bool [S,S]) — causal mask applied analytically
  const float* conv_w = (const float*)d_in[4];
  const float* conv_b = (const float*)d_in[5];
  float* out = (float*)d_out;

  // workspace carve (~25.7 MB; fits comfortably in the 192 MB L2)
  char* ws = (char*)d_ws;
  const size_t NQC = (size_t)BH_ * S_ * C_;  // 2,097,152
  const size_t NSV = (size_t)BH_ * S_;       // 65,536
  unsigned short* qc_hi = (unsigned short*)ws;
  unsigned short* qc_lo = qc_hi + NQC;
  unsigned short* kc_hi = qc_lo + NQC;
  unsigned short* kc_lo = kc_hi + NQC;
  float* q2 = (float*)(kc_lo + NQC);
  float* k2 = q2 + NSV;
  unsigned short* vt = (unsigned short*)(k2 + NSV);  // [BH][HD][S] bf16

  conv_prep<<<dim3(BH_, S_ / 64), 256, 0, stream>>>(
      query, keys, vals, conv_w, conv_b, qc_hi, qc_lo, kc_hi, kc_lo, q2, k2,
      vt);
  attn_fwd<<<(BH_ * (S_ / 16)) / 4, 128, 0, stream>>>(qc_hi, qc_lo, kc_hi,
                                                      kc_lo, q2, k2, vt, out);
}
